// APPNP_17128329576563
// MI455X (gfx1250) — compile-verified, matrix operand order checked by reading, not run on hardware
//
#include <hip/hip_runtime.h>
#include <hip/hip_bf16.h>
#include <cstdint>

// ---------------------------------------------------------------------------
// APPNP for MI455X (gfx1250): bf16 WMMA trunk + memory-bound propagation
// ---------------------------------------------------------------------------

typedef __attribute__((ext_vector_type(16))) __bf16   v16bf;
typedef __attribute__((ext_vector_type(8)))  float    v8f;
typedef __attribute__((ext_vector_type(8)))  unsigned v8u;
typedef __attribute__((ext_vector_type(4)))  unsigned v4u;

#define N_NODES   100000
#define N_EDGES   1600000
#define IN_FEATS  512
#define N_HIDDEN  1024
#define N_CLASSES 64
#define K_ITERS   10
#define ALPHA     0.1f

#if __has_builtin(__builtin_amdgcn_global_load_async_to_lds_b128) && \
    __has_builtin(__builtin_amdgcn_s_wait_asynccnt)
#define HAVE_ASYNC_LDS 1
typedef int v4i __attribute__((vector_size(16)));
typedef __attribute__((address_space(1))) void GlobalVoid;
typedef __attribute__((address_space(3))) void LdsVoid;
typedef __attribute__((address_space(1))) v4i  GlobalV4i;
typedef __attribute__((address_space(3))) v4i  LdsV4i;
#else
#define HAVE_ASYNC_LDS 0
#endif

// ------------------------- fp32 -> bf16 cast --------------------------------
__global__ void cast_f32_bf16(const float* __restrict__ in,
                              __bf16* __restrict__ out, long n) {
    long i = (long)blockIdx.x * blockDim.x + threadIdx.x;
    long stride = (long)gridDim.x * blockDim.x;
    for (; i < n; i += stride) out[i] = (__bf16)in[i];
}

// ------------------------- WMMA bf16 GEMM -----------------------------------
// C[M,Ncols] = act(A[M,K] @ B[K,Ncols] + bias), A/B bf16 row-major, acc fp32.
// Block: 256 threads = 8 waves. Tile: 128 x (NT*16). K-step 32.
// LDS layouts are chosen so BOTH fragments are two aligned ds_load_b128:
//   As[r][k]   row-major (80B row stride)
//   Bs[n][kp]  transposed + pair-packed: dword kp = {B[2kp][n], B[2kp+1][n]}
template <int NT, bool RELU, bool OUT_BF16>
__global__ __launch_bounds__(256)
void gemm_bf16_wmma(const __bf16* __restrict__ A, const __bf16* __restrict__ B,
                    const float* __restrict__ bias, void* __restrict__ Cout,
                    int M, int Kdim, int Ncols) {
    constexpr int BM = 128, BK = 32, BN = NT * 16;
    constexpr int APAD = 8;                       // pad A rows to 80B
    constexpr int BPAD = 4;                       // pad B rows to 80B (20 dwords)
    __shared__ __attribute__((aligned(16))) __bf16   As[BM][BK + APAD];
    __shared__ __attribute__((aligned(16))) unsigned Bs[BN][16 + BPAD];

    const int t    = threadIdx.x;
    const int lane = t & 31;
    const int wave = t >> 5;
    const int half = lane >> 4;                   // 0 | 1 (K-half select)
    const int l16  = lane & 15;

    const int blockRow = blockIdx.y * BM;
    const int blockCol = blockIdx.x * BN;
    const int kbase    = half * 8;
    const int arow     = wave * 16 + l16;

    v8f acc[NT] = {};

    for (int k0 = 0; k0 < Kdim; k0 += BK) {
        // ---- stage A tile (BM x BK)
#if HAVE_ASYNC_LDS
        // async copy, 16B per lane-slot; clamp OOB rows (they only feed OOB C
        // rows which are never stored)
        #pragma unroll
        for (int it = 0; it < (BM * BK * 2) / (256 * 16); ++it) {
            int slot = t + it * 256;
            int r  = slot >> 2;
            int c8 = (slot & 3) * 8;              // 8 bf16 = 16B
            int gr = blockRow + r;
            gr = gr < M ? gr : (M - 1);
            __builtin_amdgcn_global_load_async_to_lds_b128(
                (GlobalV4i*)(GlobalVoid*)(void*)(const_cast<__bf16*>(A) +
                                                 (long)gr * Kdim + k0 + c8),
                (LdsV4i*)(LdsVoid*)(void*)(&As[r][c8]), 0, 0);
        }
#else
        #pragma unroll
        for (int it = 0; it < (BM * BK) / (256 * 4); ++it) {
            int slot = t + it * 256;
            int r = slot >> 3;
            int c = (slot & 7) * 4;
            int gr = blockRow + r;
            unsigned long long v = 0ull;
            if (gr < M)
                v = *(const unsigned long long*)(A + (long)gr * Kdim + k0 + c);
            *(unsigned long long*)(&As[r][c]) = v;
        }
#endif
        // ---- stage B tile pair-packed transposed: Bs[n][kp]
        #pragma unroll
        for (int it = 0; it < (BN * 16) / 256; ++it) {
            int idx = t + it * 256;
            int kp = idx / BN;                    // n fast -> coalesced global
            int n  = idx % BN;
            long gb = (long)(k0 + 2 * kp) * Ncols + blockCol + n;
            unsigned lo = *(const unsigned short*)(B + gb);
            unsigned hi = *(const unsigned short*)(B + gb + Ncols);
            Bs[n][kp] = lo | (hi << 16);
        }
#if HAVE_ASYNC_LDS
        __builtin_amdgcn_s_wait_asynccnt(0);
#endif
        __syncthreads();

        // ---- A fragment: two b128 reads (K kbase..+7, kbase+16..+23)
        v16bf afrag;
        {
            v4u lo = *(const v4u*)(&As[arow][kbase]);
            v4u hi = *(const v4u*)(&As[arow][kbase + 16]);
            v8u aw;
            #pragma unroll
            for (int j = 0; j < 4; ++j) { aw[j] = lo[j]; aw[j + 4] = hi[j]; }
            afrag = __builtin_bit_cast(v16bf, aw);
        }
        // ---- NT WMMA tiles across N; B fragment is also two b128 reads
        #pragma unroll
        for (int nt = 0; nt < NT; ++nt) {
            int n = nt * 16 + l16;
            v4u blo = *(const v4u*)(&Bs[n][half * 4]);
            v4u bhi = *(const v4u*)(&Bs[n][8 + half * 4]);
            v8u bw;
            #pragma unroll
            for (int j = 0; j < 4; ++j) { bw[j] = blo[j]; bw[j + 4] = bhi[j]; }
            v16bf bfrag = __builtin_bit_cast(v16bf, bw);
            acc[nt] = __builtin_amdgcn_wmma_f32_16x16x32_bf16(
                false, afrag, false, bfrag, (short)0, acc[nt], false, false);
        }
        __syncthreads();
    }

    // ---- epilogue: bias (+ReLU) then bf16 or fp32 store
    #pragma unroll
    for (int nt = 0; nt < NT; ++nt) {
        int gcol = blockCol + nt * 16 + l16;
        float bv = bias[gcol];
        #pragma unroll
        for (int vi = 0; vi < 8; ++vi) {
            int grow = blockRow + wave * 16 + vi + half * 8;
            if (grow < M) {
                float val = acc[nt][vi] + bv;
                if (RELU) val = val > 0.f ? val : 0.f;
                long o = (long)grow * Ncols + gcol;
                if (OUT_BF16) ((__bf16*)Cout)[o] = (__bf16)val;
                else          ((float*)Cout)[o]  = val;
            }
        }
    }
}

// ------------------------- propagation kernels ------------------------------
__global__ void zero_f32(float* __restrict__ p, long n) {
    long i = (long)blockIdx.x * blockDim.x + threadIdx.x;
    long stride = (long)gridDim.x * blockDim.x;
    for (; i < n; i += stride) p[i] = 0.f;
}

__global__ void degree_kernel(const int* __restrict__ dst,
                              float* __restrict__ deg, int E) {
    int e = blockIdx.x * blockDim.x + threadIdx.x;
    if (e < E) atomicAdd(&deg[dst[e]], 1.0f);
}

__global__ void norm_kernel(float* __restrict__ deg, int n) {
    int i = blockIdx.x * blockDim.x + threadIdx.x;
    if (i < n) {
        float d = deg[i];
        d = d < 1.f ? 1.f : d;
        deg[i] = rsqrtf(d);            // in-place deg -> norm
    }
}

// one thread per (edge, class): agg[dst] += h[src] * norm[src]
__global__ void scatter_kernel(const float* __restrict__ h,
                               const int* __restrict__ src,
                               const int* __restrict__ dst,
                               const float* __restrict__ norm,
                               float* __restrict__ agg, int E) {
    long tid = (long)blockIdx.x * blockDim.x + threadIdx.x;
    if (tid >= (long)E * N_CLASSES) return;
    int e = (int)(tid >> 6);
    int c = (int)(tid & 63);
    int s = src[e], d = dst[e];
    float v = h[(long)s * N_CLASSES + c] * norm[s];
    atomicAdd(&agg[(long)d * N_CLASSES + c], v);
}

// out = (1-alpha) * agg * norm[node] + alpha * h0   (out may alias agg)
__global__ void combine_kernel(const float* __restrict__ agg,
                               const float* __restrict__ h0,
                               const float* __restrict__ norm,
                               float* __restrict__ out, int n) {
    long tid = (long)blockIdx.x * blockDim.x + threadIdx.x;
    if (tid >= (long)n * N_CLASSES) return;
    int node = (int)(tid >> 6);
    out[tid] = (1.0f - ALPHA) * agg[tid] * norm[node] + ALPHA * h0[tid];
}

// ------------------------- launch -------------------------------------------
extern "C" void kernel_launch(void* const* d_in, const int* in_sizes, int n_in,
                              void* d_out, int out_size, void* d_ws, size_t ws_size,
                              hipStream_t stream) {
    const float* features = (const float*)d_in[0];
    const int*   src      = (const int*)d_in[1];
    const int*   dst      = (const int*)d_in[2];
    const float* W0 = (const float*)d_in[3];
    const float* b0 = (const float*)d_in[4];
    const float* W1 = (const float*)d_in[5];
    const float* b1 = (const float*)d_in[6];
    const float* W2 = (const float*)d_in[7];
    const float* b2 = (const float*)d_in[8];

    // workspace carve-up
    char* ws = (char*)d_ws;
    size_t off = 0;
    auto carve = [&](size_t bytes) {
        void* p = ws + off;
        off += (bytes + 255) & ~(size_t)255;
        return p;
    };
    __bf16* featB = (__bf16*)carve((size_t)N_NODES * IN_FEATS * 2);
    __bf16* h1B   = (__bf16*)carve((size_t)N_NODES * N_HIDDEN * 2);
    __bf16* h2B   = (__bf16*)carve((size_t)N_NODES * N_HIDDEN * 2);
    __bf16* W0B   = (__bf16*)carve((size_t)IN_FEATS * N_HIDDEN * 2);
    __bf16* W1B   = (__bf16*)carve((size_t)N_HIDDEN * N_HIDDEN * 2);
    __bf16* W2B   = (__bf16*)carve((size_t)N_HIDDEN * N_CLASSES * 2);
    float*  h0    = (float*)carve((size_t)N_NODES * N_CLASSES * 4);
    float*  bufA  = (float*)carve((size_t)N_NODES * N_CLASSES * 4);
    float*  bufB  = (float*)carve((size_t)N_NODES * N_CLASSES * 4);
    float*  nrm   = (float*)carve((size_t)N_NODES * 4);

    const int THR = 256;

    // 1) fp32 -> bf16 casts
    cast_f32_bf16<<<4096, THR, 0, stream>>>(features, featB, (long)N_NODES * IN_FEATS);
    cast_f32_bf16<<<1024, THR, 0, stream>>>(W0, W0B, (long)IN_FEATS * N_HIDDEN);
    cast_f32_bf16<<<1024, THR, 0, stream>>>(W1, W1B, (long)N_HIDDEN * N_HIDDEN);
    cast_f32_bf16<<<256,  THR, 0, stream>>>(W2, W2B, (long)N_HIDDEN * N_CLASSES);

    // 2) MLP trunk (WMMA bf16, f32 accumulate)
    dim3 g0(N_HIDDEN / 128, (N_NODES + 127) / 128);
    gemm_bf16_wmma<8, true, true><<<g0, THR, 0, stream>>>(
        featB, W0B, b0, h1B, N_NODES, IN_FEATS, N_HIDDEN);
    gemm_bf16_wmma<8, true, true><<<g0, THR, 0, stream>>>(
        h1B, W1B, b1, h2B, N_NODES, N_HIDDEN, N_HIDDEN);
    dim3 g2(N_CLASSES / 64, (N_NODES + 127) / 128);
    gemm_bf16_wmma<4, false, false><<<g2, THR, 0, stream>>>(
        h2B, W2B, b2, h0, N_NODES, N_HIDDEN, N_CLASSES);

    // 3) symmetric normalization: deg -> rsqrt(max(deg,1)) in-place
    zero_f32<<<512, THR, 0, stream>>>(nrm, N_NODES);
    degree_kernel<<<(N_EDGES + THR - 1) / THR, THR, 0, stream>>>(dst, nrm, N_EDGES);
    norm_kernel<<<(N_NODES + THR - 1) / THR, THR, 0, stream>>>(nrm, N_NODES);

    // 4) K propagation steps (ping-pong bufA/bufB; h0 stays pristine)
    const long nh   = (long)N_NODES * N_CLASSES;
    const long ne   = (long)N_EDGES * N_CLASSES;
    const int  gSc  = (int)((ne + THR - 1) / THR);
    const int  gCm  = (int)((nh + THR - 1) / THR);
    const float* hcur = h0;
    float* bufs[2] = {bufA, bufB};
    for (int k = 0; k < K_ITERS; ++k) {
        float* agg = bufs[k & 1];
        zero_f32<<<2048, THR, 0, stream>>>(agg, nh);
        scatter_kernel<<<gSc, THR, 0, stream>>>(hcur, src, dst, nrm, agg, N_EDGES);
        float* outp = (k == K_ITERS - 1) ? (float*)d_out : agg;
        combine_kernel<<<gCm, THR, 0, stream>>>(agg, h0, nrm, outp, N_NODES);
        hcur = outp;
    }
}
